// ClusterMemory_37366215475660
// MI455X (gfx1250) — matrix operand order repaired and checked by reading, not compiled, not run on hardware
//
#include <hip/hip_runtime.h>
#include <hip/hip_bf16.h>

// ---------------- problem constants ----------------
#define B_SZ   1024
#define D_SZ   2048
#define N_SZ   16384
#define TEMP_F 0.05f
#define LAM2   0.5f
#define MU_F   1.0f
#define EPS_N  1e-12f

// ---------------- WMMA / TDM types ----------------
typedef __attribute__((ext_vector_type(16))) __bf16        v16bf;
typedef __attribute__((ext_vector_type(8)))  float         v8f;
typedef __attribute__((ext_vector_type(4)))  unsigned int  v4u;
typedef __attribute__((ext_vector_type(8)))  int           v8i;
typedef __attribute__((ext_vector_type(4)))  int           v4i;

union FragU {
    v16bf v;
    uint4 q[2];
};

// fp32 -> bf16 round-to-nearest-even
__device__ __forceinline__ unsigned short f2bf(float f) {
    unsigned int u = __float_as_uint(f);
    unsigned int r = (u + 0x7FFFu + ((u >> 16) & 1u)) >> 16;
    return (unsigned short)r;
}

// ---------------- block reductions (256 threads) ----------------
__device__ __forceinline__ float blockReduceSum(float v, float* sm) {
    int tid = threadIdx.x;
    sm[tid] = v; __syncthreads();
    for (int s = 128; s > 0; s >>= 1) {
        if (tid < s) sm[tid] += sm[tid + s];
        __syncthreads();
    }
    float r = sm[0]; __syncthreads();
    return r;
}

__device__ __forceinline__ float blockReduceMax(float v, float* sm) {
    int tid = threadIdx.x;
    sm[tid] = v; __syncthreads();
    for (int s = 128; s > 0; s >>= 1) {
        if (tid < s) sm[tid] = fmaxf(sm[tid], sm[tid + s]);
        __syncthreads();
    }
    float r = sm[0]; __syncthreads();
    return r;
}

// ---------------- 0) zero accumulators ----------------
__global__ void k_zero(float* acc) {
    if (threadIdx.x < 16) acc[threadIdx.x] = 0.0f;
}

// ---------------- 1) l2norm(student)->bf16, MSE(student_n, teacher_n) ----------------
// ld = sum_d mean_b (xn - tn)^2 = ||Xn - Tn||_F^2 / B  (divide by B in combine)
__global__ void __launch_bounds__(256) k_norm_mse(const float* __restrict__ x,
                                                  const float* __restrict__ t,
                                                  unsigned short* __restrict__ xbf,
                                                  float* __restrict__ ld_acc) {
    __shared__ float sm[256];
    const int b = blockIdx.x, tid = threadIdx.x;
    const float* xr = x + (size_t)b * D_SZ;
    const float* tr = t + (size_t)b * D_SZ;
    float sx = 0.0f, st = 0.0f;
    for (int i = tid; i < D_SZ; i += 256) {
        float a = xr[i]; sx += a * a;
        float c = tr[i]; st += c * c;
    }
    sx = blockReduceSum(sx, sm);
    st = blockReduceSum(st, sm);
    const float inx  = 1.0f / fmaxf(sqrtf(sx), EPS_N);
    const float int_ = 1.0f / fmaxf(sqrtf(st), EPS_N);
    unsigned short* xo = xbf + (size_t)b * D_SZ;
    float sq = 0.0f;
    for (int i = tid; i < D_SZ; i += 256) {
        float xn = xr[i] * inx;
        float tn = tr[i] * int_;
        xo[i] = f2bf(xn);
        float d = xn - tn;
        sq += d * d;
    }
    sq = blockReduceSum(sq, sm);
    if (tid == 0) atomicAdd(ld_acc, sq);
}

// ---------------- 2) bank fp32 -> bf16, per-row squared norm ----------------
__global__ void __launch_bounds__(256) k_bank(const float* __restrict__ f,
                                              unsigned short* __restrict__ fbf,
                                              float* __restrict__ f2) {
    __shared__ float sm[256];
    const int n = blockIdx.x, tid = threadIdx.x;
    const float* fr = f + (size_t)n * D_SZ;
    unsigned short* fo = fbf + (size_t)n * D_SZ;
    float s = 0.0f;
    for (int i = tid; i < D_SZ; i += 256) {
        float a = fr[i];
        s += a * a;
        fo[i] = f2bf(a);
    }
    s = blockReduceSum(s, sm);
    if (tid == 0) f2[n] = s;
}

// ---------------- 3) bf16 WMMA GEMM with TDM-fed LDS: S = A[M,K] * Bank[N,K]^T ----
#define BM 64
#define BN 128
#define BK 64
#define LDSK 72   // 128B data + 16B pad per row (matches D# pad: 4 DW after 32 DW)

// Issue one 2-D TDM tile load: rows x BK bf16 elements, row stride D_SZ elements,
// LDS rows padded to LDSK*2 bytes via D# pad_enable.
__device__ __forceinline__ void tdm_load_tile(unsigned lds_off,
                                              const unsigned short* gptr,
                                              unsigned tile_rows) {
    unsigned long long ga = (unsigned long long)(const void*)gptr;
    // Group 0: count=1 | lds_addr | global_addr[56:0] | type=2
    v4u g0 = { 1u,
               lds_off,
               (unsigned)ga,
               (unsigned)((ga >> 32) & 0x01FFFFFFull) | (2u << 30) };
    // Group 1: data_size=2B (1<<16), pad_enable (1<<20),
    //          pad_interval=4 -> pad after 32 DWORDs, pad_amount=3 -> 4 DWORDs
    const unsigned w0 = (1u << 16) | (1u << 20) | (4u << 22) | (3u << 25);
    v8i g1 = { (int)w0,
               (int)(((unsigned)D_SZ & 0xFFFFu) << 16),                      // tensor_dim0 lo
               (int)((((unsigned)D_SZ >> 16) & 0xFFFFu)
                     | ((tile_rows & 0xFFFFu) << 16)),                       // dim0 hi | tensor_dim1 lo
               (int)(((unsigned)BK) << 16),                                  // tensor_dim1 hi=0 | tile_dim0
               (int)(tile_rows & 0xFFFFu),                                   // tile_dim1 | tile_dim2=0
               (int)(unsigned)D_SZ,                                          // tensor_dim0_stride lo32
               0, 0 };
    v4i gz4 = { 0, 0, 0, 0 };
    v8i gz8 = { 0, 0, 0, 0, 0, 0, 0, 0 };
    __builtin_amdgcn_tensor_load_to_lds(g0, g1, gz4, gz4, gz8, 0);
}

__device__ __forceinline__ unsigned lds_byte_off(const void* p) {
    // generic LDS pointer: addr[31:0] is the LDS byte offset (aperture in high bits)
    return (unsigned)(unsigned long long)p;
}

__global__ void __launch_bounds__(256) k_gemm(const unsigned short* __restrict__ A,
                                              const unsigned short* __restrict__ Bm,
                                              float* __restrict__ S) {
    __shared__ __align__(16) unsigned short Al[2][BM * LDSK];
    __shared__ __align__(16) unsigned short Bl[2][BN * LDSK];

    const int tid  = threadIdx.x;
    const int lane = tid & 31;
    const int wave = tid >> 5;
    const int wm = (wave & 3) * 16;   // 4 waves over M
    const int wn = (wave >> 2) * 64;  // 2 waves over N
    const int m0 = blockIdx.y * BM;
    const int n0 = blockIdx.x * BN;

    const v8f vzero = {0.f, 0.f, 0.f, 0.f, 0.f, 0.f, 0.f, 0.f};
    v8f acc0 = vzero, acc1 = vzero, acc2 = vzero, acc3 = vzero;

    const int hi   = lane >> 4;        // 0: lanes 0-15, 1: lanes 16-31
    const int aoff = hi * 16;          // A frag: K 0..15 / 16..31
    const int boff = hi * 16;          // B frag: K 0..15 / 16..31
    const int arow = wm + (lane & 15);
    const int bcol = wn + (lane & 15);
    const bool leader = (wave == 0);

    const unsigned short* Ag = A  + (size_t)m0 * D_SZ;
    const unsigned short* Bg = Bm + (size_t)n0 * D_SZ;

    if (leader) {   // prologue: fill buffer 0 via TDM
        tdm_load_tile(lds_byte_off(&Al[0][0]), Ag, BM);
        tdm_load_tile(lds_byte_off(&Bl[0][0]), Bg, BN);
    }

    const int nP = D_SZ / BK;  // 32 K-panels
    for (int p = 0; p < nP; ++p) {
        const int cur = p & 1;
        if (leader) {
            if (p + 1 < nP) {
                // issue next panel into the other buffer, then retire current panel
                tdm_load_tile(lds_byte_off(&Al[cur ^ 1][0]), Ag + (p + 1) * BK, BM);
                tdm_load_tile(lds_byte_off(&Bl[cur ^ 1][0]), Bg + (p + 1) * BK, BN);
                __builtin_amdgcn_s_wait_tensorcnt(2);  // in-order: current panel done
            } else {
                __builtin_amdgcn_s_wait_tensorcnt(0);
            }
        }
        __syncthreads();  // current buffer visible to all waves

        const unsigned short* Ab = &Al[cur][0];
        const unsigned short* Bb = &Bl[cur][0];
#pragma unroll
        for (int kk = 0; kk < BK; kk += 32) {
            FragU a;
            a.q[0] = *(const uint4*)&Ab[arow * LDSK + kk + aoff];
            a.q[1] = *(const uint4*)&Ab[arow * LDSK + kk + 8 + aoff];
            FragU b0, b1, b2, b3;
            b0.q[0] = *(const uint4*)&Bb[(bcol +  0) * LDSK + kk + boff];
            b0.q[1] = *(const uint4*)&Bb[(bcol +  0) * LDSK + kk + boff + 8];
            b1.q[0] = *(const uint4*)&Bb[(bcol + 16) * LDSK + kk + boff];
            b1.q[1] = *(const uint4*)&Bb[(bcol + 16) * LDSK + kk + boff + 8];
            b2.q[0] = *(const uint4*)&Bb[(bcol + 32) * LDSK + kk + boff];
            b2.q[1] = *(const uint4*)&Bb[(bcol + 32) * LDSK + kk + boff + 8];
            b3.q[0] = *(const uint4*)&Bb[(bcol + 48) * LDSK + kk + boff];
            b3.q[1] = *(const uint4*)&Bb[(bcol + 48) * LDSK + kk + boff + 8];
            acc0 = __builtin_amdgcn_wmma_f32_16x16x32_bf16(false, a.v, false, b0.v, (short)0, acc0, false, false);
            acc1 = __builtin_amdgcn_wmma_f32_16x16x32_bf16(false, a.v, false, b1.v, (short)0, acc1, false, false);
            acc2 = __builtin_amdgcn_wmma_f32_16x16x32_bf16(false, a.v, false, b2.v, (short)0, acc2, false, false);
            acc3 = __builtin_amdgcn_wmma_f32_16x16x32_bf16(false, a.v, false, b3.v, (short)0, acc3, false, false);
        }
        __syncthreads();  // all reads done before TDM overwrites this buffer
    }

    // C/D layout: VGPR r -> M = r + (lane>=16 ? 8 : 0); N = lane & 15
    const int srow0 = m0 + wm + (hi ? 8 : 0);
    const int scol0 = n0 + wn + (lane & 15);
    float* p0 = S + (size_t)srow0 * N_SZ + scol0;
#pragma unroll
    for (int r = 0; r < 8; ++r) {
        p0[(size_t)r * N_SZ +  0] = acc0[r];
        p0[(size_t)r * N_SZ + 16] = acc1[r];
        p0[(size_t)r * N_SZ + 32] = acc2[r];
        p0[(size_t)r * N_SZ + 48] = acc3[r];
    }
}

// ---------------- 4) per-row epilogue: both cross-entropies ----------------
// logits1 = S/TEMP ; d = sqrt(max(1 + f2 - 2S, 0)) ; p = softmax(d) ; logits2 = p
__global__ void __launch_bounds__(256) k_epi(const float* __restrict__ S,
                                             const float* __restrict__ f2,
                                             const int* __restrict__ tgt,
                                             float* __restrict__ ce1_acc,
                                             float* __restrict__ ce2_acc) {
    __shared__ float sm[256];
    const int b = blockIdx.x, tid = threadIdx.x;
    const float* Sr = S + (size_t)b * N_SZ;
    const float itemp = 1.0f / TEMP_F;

    float ms = -3.0e38f, md = -3.0e38f;
    for (int i = tid; i < N_SZ; i += 256) {
        float s = Sr[i];
        ms = fmaxf(ms, s);
        float d = sqrtf(fmaxf(1.0f + f2[i] - 2.0f * s, 0.0f));
        md = fmaxf(md, d);
    }
    ms = blockReduceMax(ms, sm);
    md = blockReduceMax(md, sm);

    float ssum = 0.0f, zd = 0.0f;
    for (int i = tid; i < N_SZ; i += 256) {
        float s = Sr[i];
        ssum += __expf((s - ms) * itemp);
        float d = sqrtf(fmaxf(1.0f + f2[i] - 2.0f * s, 0.0f));
        zd += __expf(d - md);
    }
    ssum = blockReduceSum(ssum, sm);
    zd   = blockReduceSum(zd, sm);
    const float izd = 1.0f / zd;

    float psum = 0.0f;   // p in [0,1] -> exp(p) stable without max-shift
    for (int i = tid; i < N_SZ; i += 256) {
        float s = Sr[i];
        float d = sqrtf(fmaxf(1.0f + f2[i] - 2.0f * s, 0.0f));
        psum += __expf(__expf(d - md) * izd);
    }
    psum = blockReduceSum(psum, sm);

    if (tid == 0) {
        const int t = tgt[b];
        const float st = Sr[t];
        const float dt = sqrtf(fmaxf(1.0f + f2[t] - 2.0f * st, 0.0f));
        const float pt = __expf(dt - md) * izd;
        const float ce1 = (ms * itemp + __logf(ssum)) - st * itemp;
        const float ce2 = __logf(psum) - pt;
        atomicAdd(ce1_acc, ce1);
        atomicAdd(ce2_acc, ce2);
    }
}

// ---------------- 5) combine ----------------
__global__ void k_combine(const float* __restrict__ acc, float* __restrict__ out) {
    const float invB = 1.0f / (float)B_SZ;
    float l0 = (acc[0] + acc[3]) * invB + MU_F * acc[6] * invB;
    float l1 = (acc[1] + acc[4]) * invB + MU_F * acc[7] * invB;
    float l2 = (acc[2] + acc[5]) * invB + MU_F * acc[8] * invB;
    out[0] = (1.0f - LAM2) * l0 + LAM2 * l1 + LAM2 * l2;
}

// ---------------- host launch ----------------
extern "C" void kernel_launch(void* const* d_in, const int* in_sizes, int n_in,
                              void* d_out, int out_size, void* d_ws, size_t ws_size,
                              hipStream_t stream) {
    (void)in_sizes; (void)n_in; (void)out_size; (void)ws_size;

    const float* x[3]    = {(const float*)d_in[0], (const float*)d_in[1], (const float*)d_in[2]};
    const float* t[3]    = {(const float*)d_in[3], (const float*)d_in[4], (const float*)d_in[5]};
    const int*   targets = (const int*)d_in[6];
    // d_in[7] = epoch (unused)
    const float* feat[3] = {(const float*)d_in[8], (const float*)d_in[9], (const float*)d_in[10]};

    char* ws = (char*)d_ws;
    float* acc = (float*)ws;                 // [0..2]=ce1, [3..5]=ce2, [6..8]=ld
    size_t off = 256;

    unsigned short* xbf[3];
    for (int i = 0; i < 3; ++i) { xbf[i] = (unsigned short*)(ws + off); off += (size_t)B_SZ * D_SZ * 2; }
    unsigned short* fbf[3];
    for (int i = 0; i < 3; ++i) { fbf[i] = (unsigned short*)(ws + off); off += (size_t)N_SZ * D_SZ * 2; }
    float* f2p[3];
    for (int i = 0; i < 3; ++i) { f2p[i] = (float*)(ws + off); off += (size_t)N_SZ * 4; }
    float* S = (float*)(ws + off);           // reused across branches (stream-serial)

    k_zero<<<1, 32, 0, stream>>>(acc);

    const dim3 ggrid(N_SZ / BN, B_SZ / BM);
    for (int i = 0; i < 3; ++i) {
        k_norm_mse<<<B_SZ, 256, 0, stream>>>(x[i], t[i], xbf[i], acc + 6 + i);
        k_bank<<<N_SZ, 256, 0, stream>>>(feat[i], fbf[i], f2p[i]);
        k_gemm<<<ggrid, 256, 0, stream>>>(xbf[i], fbf[i], S);
        k_epi<<<B_SZ, 256, 0, stream>>>(S, f2p[i], targets, acc + i, acc + 3 + i);
    }

    k_combine<<<1, 1, 0, stream>>>(acc, (float*)d_out);
}
